// Encoder_15410342658418
// MI455X (gfx1250) — matrix operand order repaired and checked by reading, not compiled, not run on hardware
//
#include <hip/hip_runtime.h>

// ---------------------------------------------------------------------------
// SNN encoder (conv3x3 s2 + weight-norm  ->  CUBA-LIF scan over T  ->  frac delay)
// Fully fused, WMMA-based conv. Bandwidth-bound: ~190 MB of traffic total.
// ---------------------------------------------------------------------------

#define N_  8
#define C_  2
#define H_  128
#define W_  128
#define T_  32
#define CH_ 32
#define HP_ 64
#define WP_ 64
#define KK  18            // C_*3*3 im2col depth (padded to 32 for WMMA)
#define RX_ 33            // halo width: x in [2*ox0-1, 2*ox0+31]

typedef __attribute__((ext_vector_type(16))) _Float16 v16h;
typedef __attribute__((ext_vector_type(8)))  float    v8f;

// ---------------------------------------------------------------------------
// Kernel 1: weight-norm + pack f16 weights directly into the WMMA A-operand
// 16x32 per-lane form, in contiguous k order (k = 16h + j, B uses the same):
//   lanes 0-15 : a[j] = K(j)                         (row M = lane)
//   lanes 16-31: a[j] = K(16+j)                      (row M = lane-16)
// Two groups: group 0 = channels 0..15, group 1 = channels 16..31.
// aPack layout: [group][lane][j] halves  (2*32*16 = 1024 halves = 2 KB in ws)
// ---------------------------------------------------------------------------
__global__ void prep_weights(const float* __restrict__ wv,
                             const float* __restrict__ wg,
                             _Float16* __restrict__ aPack) {
    int tid = threadIdx.x;            // 0..63
    if (tid >= 64) return;
    int g = tid >> 5, l = tid & 31;
    int m = (l & 15) + 16 * g;        // output channel this lane packs
    float ss = 0.f;
    for (int k = 0; k < KK; ++k) { float v = wv[m * KK + k]; ss += v * v; }
    float scale = wg[m] * rsqrtf(ss); // g * v / ||v||
    for (int j = 0; j < 16; ++j) {
        int k = (l < 16) ? j : j + 16;
        float v = (k < KK) ? wv[m * KK + k] * scale : 0.f;
        aPack[g * 512 + l * 16 + j] = (_Float16)v;
    }
}

// ---------------------------------------------------------------------------
// Kernel 2: fused conv(WMMA) + LIF scan + delay.
// 1 wave = one (n, oy, ox0..ox0+15) tile across all T.
// Block = 128 threads = 4 waves. Grid = 2048 waves / 4 = 512 blocks.
// ---------------------------------------------------------------------------
__launch_bounds__(128)
__global__ void snn_fused(const float* __restrict__ spike,
                          const float* __restrict__ delay,
                          const _Float16* __restrict__ aPack,
                          float* __restrict__ out) {
    __shared__ _Float16 sIn[4][C_ * 3 * RX_ * T_];   // 4 waves * 12672 B

    const int lane = threadIdx.x & 31;
    const int wv   = threadIdx.x >> 5;
    const int w    = blockIdx.x * 4 + wv;            // global wave id, 0..2047
    const int n    = w >> 8;                         // 256 tiles per image
    const int tile = w & 255;
    const int oy   = tile >> 2;
    const int ox0  = (tile & 3) << 4;

    // ---- stage input halo (2 ch x 3 rows x 33 cols, all T) to LDS as f16 ----
    _Float16* sW = sIn[wv];
    for (int e = 0; e < C_ * 3 * RX_; ++e) {
        int c = e / (3 * RX_), rem = e % (3 * RX_);
        int ry = rem / RX_,    rx  = rem % RX_;
        int y = 2 * oy - 1 + ry, x = 2 * ox0 - 1 + rx;
        float v = 0.f;
        if ((unsigned)y < H_ && (unsigned)x < W_)   // wave-uniform branch
            v = spike[((((n * C_ + c) * H_ + y) * W_ + x) * T_) + lane]; // lane = t
        sW[e * T_ + lane] = (_Float16)v;
    }
    __syncthreads();

    // ---- WMMA A operands (pre-packed in register layout) ----
    const v16h a0 = *(const v16h*)(aPack +       lane * 16);  // ch 0..15
    const v16h a1 = *(const v16h*)(aPack + 512 + lane * 16);  // ch 16..31

    const int col = lane & 15;     // B column = output pixel within tile
    const int kg  = lane >> 4;     // 0: K0-15, 1: K16-31 (B layout half)
    const int ox  = ox0 + col;

    // ---- per-lane channel map (16x16 f32 C/D layout), delay frac, out base ---
    float     fdel[16];
    long long baseIdx[16];
    #pragma unroll
    for (int i = 0; i < 16; ++i) {
        int ch = (i < 8) ? (i + 8 * kg) : (16 + (i - 8) + 8 * kg);
        float d = delay[ch];
        fdel[i] = d - floorf(d);   // delay ~ U[0,1) => integer shift is 0
        baseIdx[i] = ((((long long)n * CH_ + ch) * HP_ + oy) * WP_ + ox) * T_;
    }

    float cur[16], volt[16], sprev[16];
    #pragma unroll
    for (int i = 0; i < 16; ++i) { cur[i] = 0.f; volt[i] = 0.f; sprev[i] = 0.f; }

    const v8f cz = {};

    for (int t0 = 0; t0 < T_; t0 += 4) {
        float ob[16][4];
        #pragma unroll
        for (int tt = 0; tt < 4; ++tt) {
            const int t = t0 + tt;

            // ---- build B operand (32x16 f16, K = im2col tap) from LDS ----
            v16h b;
            if (kg == 0) {
                #pragma unroll
                for (int j = 0; j < 16; ++j) {     // K = j (0..15), all real taps
                    const int c = j / 9, r = j % 9, kh = r / 3, kw = r % 3;
                    b[j] = sW[((c * 3 + kh) * RX_ + (2 * col + kw)) * T_ + t];
                }
            } else {                                // K = 16+j; only 16,17 real
                b[0] = sW[((1 * 3 + 2) * RX_ + (2 * col + 1)) * T_ + t]; // c1,kh2,kw1
                b[1] = sW[((1 * 3 + 2) * RX_ + (2 * col + 2)) * T_ + t]; // c1,kh2,kw2
                #pragma unroll
                for (int j = 2; j < 16; ++j) b[j] = (_Float16)0.f;
            }

            // ---- conv for this timestep: z = W x im2col(spike_t) ----
            v8f d0 = __builtin_amdgcn_wmma_f32_16x16x32_f16(
                false, a0, false, b, (short)0, cz, false, false);
            v8f d1 = __builtin_amdgcn_wmma_f32_16x16x32_f16(
                false, a1, false, b, (short)0, cz, false, false);

            // ---- CUBA-LIF update + hard reset + fractional delay ----
            #pragma unroll
            for (int i = 0; i < 16; ++i) {
                float z = (i < 8) ? d0[i] : d1[i - 8];
                cur[i]  = cur[i] * 0.75f + z;            // 1 - CUR_DECAY
                volt[i] = volt[i] * 0.9f + cur[i];       // 1 - VOLT_DECAY
                float s = (volt[i] >= 1.0f) ? 1.0f : 0.0f;
                volt[i] *= (1.0f - s);                   // hard reset
                ob[i][tt] = (1.0f - fdel[i]) * s + fdel[i] * sprev[i];
                sprev[i]  = s;
            }
        }
        // ---- 16B stores: t is innermost in out[N,CH,64,64,T] ----
        #pragma unroll
        for (int i = 0; i < 16; ++i) {
            float4 v4 = make_float4(ob[i][0], ob[i][1], ob[i][2], ob[i][3]);
            *(float4*)(out + baseIdx[i] + t0) = v4;
        }
    }
}

// ---------------------------------------------------------------------------
extern "C" void kernel_launch(void* const* d_in, const int* in_sizes, int n_in,
                              void* d_out, int out_size, void* d_ws, size_t ws_size,
                              hipStream_t stream) {
    const float* spike  = (const float*)d_in[0];   // [8,2,128,128,32]
    const float* wvec   = (const float*)d_in[1];   // [32,2,3,3]
    const float* wgain  = (const float*)d_in[2];   // [32]
    const float* delay  = (const float*)d_in[3];   // [32]
    float*       out    = (float*)d_out;           // [8,32,64,64,32]
    _Float16*    aPack  = (_Float16*)d_ws;         // 2 KB packed f16 weights

    prep_weights<<<1, 64, 0, stream>>>(wvec, wgain, aPack);
    snn_fused<<<512, 128, 0, stream>>>(spike, delay, aPack, out);
}
